// ArchGVAE_46694884442155
// MI455X (gfx1250) — matrix operand order repaired and checked by reading, not compiled, not run on hardware
//
#include <hip/hip_runtime.h>
#include <hip/hip_bf16.h>

#define BATCH 65536
#define HD 128
#define ZD 32
#define OD 78
#define KP 288   // 261 padded up to a multiple of 32 for the message GEMM

typedef __attribute__((ext_vector_type(16))) _Float16 v16h;
typedef __attribute__((ext_vector_type(8)))  float    v8f;
typedef __attribute__((ext_vector_type(4)))  float    v4f;

__constant__ int c_SRC[6] = {0, 0, 1, 0, 1, 2};
__constant__ int c_DST[6] = {1, 2, 2, 3, 3, 3};

__device__ __forceinline__ float lrelu(float x) { return x > 0.0f ? x : 0.01f * x; }

__device__ __forceinline__ v8f wmma32(v16h a, v16h b, v8f c) {
    // v_wmma_f32_16x16x32_f16: D = A(16x32 f16) * B(32x16 f16) + C(16x16 f32)
    return __builtin_amdgcn_wmma_f32_16x16x32_f16(false, a, false, b, (short)0, c, false, false);
}

// A tile stored row-major [M][lda] f16 in LDS. 16-bit A 16x32 layout:
// lane m=l&15, half h=l>>4; elems 0..7 = K[h*8 .. h*8+7], elems 8..15 = K[16+h*8 ..].
__device__ __forceinline__ v16h load_a_frag(const _Float16* A, int lda, int m0, int kb, int lane) {
    int m = lane & 15, hh = (lane >> 4) & 1;
    const _Float16* p = A + (size_t)(m0 + m) * lda + kb + hh * 8;
    union { v16h v; v4f q[2]; } u;
    u.q[0] = *(const v4f*)(p);
    u.q[1] = *(const v4f*)(p + 16);
    return u.v;
}

// B (KxN) stored transposed [N][ldb] f16 in LDS. 16-bit B 32x16 layout:
// lane n=l&15, half h=l>>4; elems 0..7 = K[h*8 .. h*8+7], elems 8..15 = K[16+h*8 ..] of column n.
__device__ __forceinline__ v16h load_b_frag(const _Float16* BT, int ldb, int n0, int kb, int lane) {
    int n = lane & 15, hh = (lane >> 4) & 1;
    const _Float16* p = BT + (size_t)(n0 + n) * ldb + kb + hh * 8;
    union { v16h v; v4f q[2]; } u;
    u.q[0] = *(const v4f*)(p);
    u.q[1] = *(const v4f*)(p + 16);
    return u.v;
}

// ---------------------------------------------------------------------------
// conv0: XDIM=4 -> HD, tiny K; VALU. 2 graphs per 256-thread block.
// ---------------------------------------------------------------------------
__global__ void __launch_bounds__(256)
k_conv0(const float* __restrict__ x, const float* __restrict__ ea,
        const float* __restrict__ rw1, const float* __restrict__ rb1,
        const float* __restrict__ rw2, const float* __restrict__ rb2,
        const float* __restrict__ kw, _Float16* __restrict__ Hout)
{
    __shared__ float s_rw1[16], s_rb1[4], s_rw2[4 * HD], s_rb2[HD], s_kw[13 * HD];
    int tid = threadIdx.x;
    for (int i = tid; i < 16; i += 256) s_rw1[i] = rw1[i];
    if (tid < 4) s_rb1[tid] = rb1[tid];
    for (int i = tid; i < 4 * HD; i += 256) s_rw2[i] = rw2[i];
    if (tid < HD) s_rb2[tid] = rb2[tid];
    for (int i = tid; i < 13 * HD; i += 256) s_kw[i] = kw[i];
    __syncthreads();

    int g = blockIdx.x * 2 + (tid >> 7);
    int j = tid & 127;
    const float* xg = x + (size_t)g * 16;   // 4 nodes x 4
    const float* eg = ea + (size_t)g * 30;  // 6 edges x 5

    float t1[4][4];
#pragma unroll
    for (int n = 0; n < 4; ++n)
#pragma unroll
        for (int c = 0; c < 4; ++c) {
            float s = s_rb1[c];
#pragma unroll
            for (int k = 0; k < 4; ++k) s += xg[n * 4 + k] * s_rw1[k * 4 + c];
            t1[n][c] = s;
        }
    float out[4];
#pragma unroll
    for (int n = 0; n < 4; ++n) {
        float s = s_rb2[j];
#pragma unroll
        for (int c = 0; c < 4; ++c) s += t1[n][c] * s_rw2[c * HD + j];
        out[n] = s;
    }
#pragma unroll
    for (int e = 0; e < 6; ++e) {
        int sN = c_SRC[e], dN = c_DST[e];
        float s = 0.0f;
#pragma unroll
        for (int k = 0; k < 4; ++k) s += xg[dN * 4 + k] * s_kw[k * HD + j];
#pragma unroll
        for (int k = 0; k < 4; ++k) s += xg[sN * 4 + k] * s_kw[(4 + k) * HD + j];
#pragma unroll
        for (int k = 0; k < 5; ++k) s += eg[e * 5 + k] * s_kw[(8 + k) * HD + j];
        out[dN] += lrelu(s);
    }
#pragma unroll
    for (int n = 0; n < 4; ++n) Hout[((size_t)g * 4 + n) * HD + j] = (_Float16)out[n];
}

// ---------------------------------------------------------------------------
// resid: R = (H@rw1+rb1)@rw2+rb2 for a 64-row tile. Two chained WMMA GEMMs.
// ---------------------------------------------------------------------------
__global__ void __launch_bounds__(256)
k_resid(const _Float16* __restrict__ Hin,
        const float* __restrict__ rw1, const float* __restrict__ rb1,
        const float* __restrict__ rw2, const float* __restrict__ rb2,
        _Float16* __restrict__ Rout)
{
    extern __shared__ char smem[];
    _Float16* w1T = (_Float16*)smem;        // [128][128]
    _Float16* w2T = w1T + HD * HD;          // [128][128]
    _Float16* At  = w2T + HD * HD;          // [64][128]
    _Float16* T1  = At + 64 * HD;           // [64][128]
    __shared__ float s_rb1[HD], s_rb2[HD];

    int tid = threadIdx.x;
    size_t row0 = (size_t)blockIdx.x * 64;
    for (int i = tid; i < HD * HD; i += 256) {
        int n = i >> 7, k = i & 127;
        w1T[i] = (_Float16)rw1[k * HD + n];
        w2T[i] = (_Float16)rw2[k * HD + n];
    }
    for (int i = tid; i < 64 * HD; i += 256) At[i] = Hin[row0 * HD + i];
    if (tid < HD) { s_rb1[tid] = rb1[tid]; s_rb2[tid] = rb2[tid]; }
    __syncthreads();

    int lane = tid & 31, w = tid >> 5, hh = (lane >> 4) & 1;
    int col = w * 16 + (lane & 15);
    {   // GEMM1: T1 = At @ rw1 + rb1
        v8f acc[4];
        float b0 = s_rb1[col];
#pragma unroll
        for (int mt = 0; mt < 4; ++mt)
#pragma unroll
            for (int r = 0; r < 8; ++r) acc[mt][r] = b0;
#pragma unroll
        for (int kb = 0; kb < HD; kb += 32) {
            v16h bf = load_b_frag(w1T, HD, w * 16, kb, lane);
#pragma unroll
            for (int mt = 0; mt < 4; ++mt) {
                v16h af = load_a_frag(At, HD, mt * 16, kb, lane);
                acc[mt] = wmma32(af, bf, acc[mt]);
            }
        }
#pragma unroll
        for (int mt = 0; mt < 4; ++mt)
#pragma unroll
            for (int r = 0; r < 8; ++r)
                T1[(mt * 16 + r + 8 * hh) * HD + col] = (_Float16)acc[mt][r];
    }
    __syncthreads();
    {   // GEMM2: Rout = T1 @ rw2 + rb2
        v8f acc[4];
        float b0 = s_rb2[col];
#pragma unroll
        for (int mt = 0; mt < 4; ++mt)
#pragma unroll
            for (int r = 0; r < 8; ++r) acc[mt][r] = b0;
#pragma unroll
        for (int kb = 0; kb < HD; kb += 32) {
            v16h bf = load_b_frag(w2T, HD, w * 16, kb, lane);
#pragma unroll
            for (int mt = 0; mt < 4; ++mt) {
                v16h af = load_a_frag(T1, HD, mt * 16, kb, lane);
                acc[mt] = wmma32(af, bf, acc[mt]);
            }
        }
#pragma unroll
        for (int mt = 0; mt < 4; ++mt)
#pragma unroll
            for (int r = 0; r < 8; ++r)
                Rout[(row0 + mt * 16 + r + 8 * hh) * HD + col] = (_Float16)acc[mt][r];
    }
}

// ---------------------------------------------------------------------------
// msg: m = leaky([h[dst]|h[src]|ea] @ kw); segment-sum; Rio += agg (in place).
// 16 graphs (96 edge rows / 64 node rows) per block. K=261 padded to 288.
// ---------------------------------------------------------------------------
__global__ void __launch_bounds__(256)
k_msg(const _Float16* __restrict__ Hin, const float* __restrict__ ea,
      const float* __restrict__ kw, _Float16* __restrict__ Rio)
{
    extern __shared__ char smem[];
    _Float16* kwT = (_Float16*)smem;             // [128][288]
    _Float16* Ae  = kwT + HD * KP;               // [96][288]
    float*    Msh = (float*)(Ae + 96 * KP);      // [96][132]

    int tid = threadIdx.x;
    int g0 = blockIdx.x * 16;
    for (int i = tid; i < HD * KP; i += 256) {
        int n = i / KP, k = i - n * KP;
        kwT[i] = (_Float16)((k < 261) ? kw[(size_t)k * HD + n] : 0.0f);
    }
    for (int i = tid; i < 96 * KP; i += 256) {
        int r = i / KP, k = i - r * KP;
        int gl = r / 6, e = r - gl * 6;
        _Float16 v = (_Float16)0.0f;
        if (k < HD)            v = Hin[((size_t)(g0 + gl) * 4 + c_DST[e]) * HD + k];
        else if (k < 2 * HD)   v = Hin[((size_t)(g0 + gl) * 4 + c_SRC[e]) * HD + (k - HD)];
        else if (k < 261)      v = (_Float16)ea[((size_t)(g0 + gl) * 6 + e) * 5 + (k - 2 * HD)];
        Ae[i] = v;
    }
    __syncthreads();

    int lane = tid & 31, w = tid >> 5, hh = (lane >> 4) & 1;
    int col = w * 16 + (lane & 15);
    {
        v8f acc[6];
#pragma unroll
        for (int et = 0; et < 6; ++et)
#pragma unroll
            for (int r = 0; r < 8; ++r) acc[et][r] = 0.0f;
#pragma unroll
        for (int kb = 0; kb < KP; kb += 32) {
            v16h bf = load_b_frag(kwT, KP, w * 16, kb, lane);
#pragma unroll
            for (int et = 0; et < 6; ++et) {
                v16h af = load_a_frag(Ae, KP, et * 16, kb, lane);
                acc[et] = wmma32(af, bf, acc[et]);
            }
        }
#pragma unroll
        for (int et = 0; et < 6; ++et)
#pragma unroll
            for (int r = 0; r < 8; ++r)
                Msh[(et * 16 + r + 8 * hh) * 132 + col] = lrelu(acc[et][r]);
    }
    __syncthreads();

    // segment-sum by fixed dst pattern {1,2,2,3,3,3}; in-place residual add.
    for (int i = tid; i < 64 * HD; i += 256) {
        int r = i >> 7, c = i & 127;
        int gl = r >> 2, n = r & 3;
        const float* Mg = Msh + (gl * 6) * 132;
        float a = 0.0f;
        if (n == 1)      a = Mg[0 * 132 + c];
        else if (n == 2) a = Mg[1 * 132 + c] + Mg[2 * 132 + c];
        else if (n == 3) a = Mg[3 * 132 + c] + Mg[4 * 132 + c] + Mg[5 * 132 + c];
        size_t idx = ((size_t)(g0 + gl) * 4 + n) * HD + c;
        Rio[idx] = (_Float16)((float)Rio[idx] + a);
    }
}

// ---------------------------------------------------------------------------
// head: readout + VAE + decoder + CE/KLD partial sums. 64 graphs per block.
// ---------------------------------------------------------------------------
__global__ void __launch_bounds__(256)
k_head(const _Float16* __restrict__ H, const float* __restrict__ eps,
       const float* __restrict__ arch,
       const float* __restrict__ fc3_w, const float* __restrict__ fc3_b,
       const float* __restrict__ fc4_w, const float* __restrict__ fc4_b,
       const float* __restrict__ fc5_w, const float* __restrict__ fc5_b,
       const float* __restrict__ d1_mw, const float* __restrict__ d1_mb,
       const float* __restrict__ d1_rw, const float* __restrict__ d1_rb,
       const float* __restrict__ d2_mw, const float* __restrict__ d2_mb,
       const float* __restrict__ d2_rw, const float* __restrict__ d2_rb,
       float* __restrict__ acc)
{
    extern __shared__ char smem[];
    _Float16* hgh  = (_Float16*)smem;                 // [64][128] hg (f16)
    _Float16* w34T = hgh + 64 * HD;                   // [64][128] [fc3|fc4]^T
    float*    muv  = (float*)(w34T + 64 * HD);        // [64][64] mu|logvar -> z
    _Float16* Hgh  = (_Float16*)(muv + 64 * 64);      // [64][128] tanh out
    _Float16* d1T  = Hgh + 64 * HD;                   // [256][128] [d1_mw|d1_rw]^T
    _Float16* h1h  = d1T + 256 * HD;                  // [64][128]
    _Float16* d2T  = h1h + 64 * HD;                   // [160][128] [d2_mw|d2_rw]^T (N pad 80)
    float*    pred = (float*)(d2T + 160 * HD);        // [64][80]
    __shared__ float s_f3b[32], s_f4b[32], s_f5b[HD], s_m1b[HD], s_r1b[HD], s_m2b[OD], s_r2b[OD];
    __shared__ float redA[256], redB[256];

    int tid = threadIdx.x;
    int g0 = blockIdx.x * 64;

    for (int i = tid; i < 64 * HD; i += 256) {
        int r = i >> 7, c = i & 127;
        float s = 0.0f;
#pragma unroll
        for (int n = 0; n < 4; ++n) s += (float)H[((size_t)(g0 + r) * 4 + n) * HD + c];
        hgh[i] = (_Float16)s;
    }
    for (int i = tid; i < 64 * HD; i += 256) {
        int n = i >> 7, k = i & 127;
        w34T[i] = (_Float16)((n < 32) ? fc3_w[k * 32 + n] : fc4_w[k * 32 + (n - 32)]);
    }
    for (int i = tid; i < 256 * HD; i += 256) {
        int n = i >> 7, k = i & 127;
        d1T[i] = (_Float16)((n < HD) ? d1_mw[k * HD + n] : d1_rw[k * HD + (n - HD)]);
    }
    for (int i = tid; i < 160 * HD; i += 256) {
        int n = i >> 7, k = i & 127;
        float v;
        if (n < 80) v = (n < OD) ? d2_mw[k * OD + n] : 0.0f;
        else { int n2 = n - 80; v = (n2 < OD) ? d2_rw[k * OD + n2] : 0.0f; }
        d2T[i] = (_Float16)v;
    }
    if (tid < 32) { s_f3b[tid] = fc3_b[tid]; s_f4b[tid] = fc4_b[tid]; }
    if (tid < HD) { s_f5b[tid] = fc5_b[tid]; s_m1b[tid] = d1_mb[tid]; s_r1b[tid] = d1_rb[tid]; }
    if (tid < OD) { s_m2b[tid] = d2_mb[tid]; s_r2b[tid] = d2_rb[tid]; }
    __syncthreads();

    int lane = tid & 31, w = tid >> 5, hh = (lane >> 4) & 1;

    // [mu|logvar] = hg @ [fc3|fc4] + b    (64x64, 16 tiles, 2 per wave)
#pragma unroll
    for (int ti = 0; ti < 2; ++ti) {
        int t = w * 2 + ti;
        int mt = t >> 2, nt = t & 3;
        int col = nt * 16 + (lane & 15);
        float b0 = (col < 32) ? s_f3b[col] : s_f4b[col - 32];
        v8f c;
#pragma unroll
        for (int r = 0; r < 8; ++r) c[r] = b0;
#pragma unroll
        for (int kb = 0; kb < HD; kb += 32) {
            v16h bf = load_b_frag(w34T, HD, nt * 16, kb, lane);
            v16h af = load_a_frag(hgh, HD, mt * 16, kb, lane);
            c = wmma32(af, bf, c);
        }
#pragma unroll
        for (int r = 0; r < 8; ++r) muv[(mt * 16 + r + 8 * hh) * 64 + col] = c[r];
    }
    __syncthreads();

    // z = eps*0.01*exp(0.5*lv)+mu (overwrite mu slot); accumulate KLD partial
    float kl = 0.0f;
    for (int i = tid; i < 64 * ZD; i += 256) {
        int g = i >> 5, d = i & 31;
        float mu = muv[g * 64 + d], lv = muv[g * 64 + 32 + d];
        kl += 1.0f + lv - mu * mu - __expf(lv);
        float z = eps[(size_t)(g0 + g) * ZD + d] * 0.01f * __expf(0.5f * lv) + mu;
        muv[g * 64 + d] = z;
    }
    __syncthreads();

    // Hg = tanh(z @ fc5 + b)   (K=32; scalar)
    for (int i = tid; i < 64 * HD; i += 256) {
        int g = i >> 7, c = i & 127;
        float s = s_f5b[c];
#pragma unroll
        for (int d = 0; d < ZD; ++d) s += muv[g * 64 + d] * fc5_w[d * HD + c];
        Hgh[i] = (_Float16)tanhf(s);
    }
    __syncthreads();

    // h1 = leaky(Hg@d1_mw+mb) + Hg@d1_rw+rb
    {
        int col = w * 16 + (lane & 15);
#pragma unroll
        for (int mt = 0; mt < 4; ++mt) {
            v8f cm, cr;
#pragma unroll
            for (int r = 0; r < 8; ++r) { cm[r] = s_m1b[col]; cr[r] = s_r1b[col]; }
#pragma unroll
            for (int kb = 0; kb < HD; kb += 32) {
                v16h af = load_a_frag(Hgh, HD, mt * 16, kb, lane);
                v16h bm = load_b_frag(d1T, HD, w * 16, kb, lane);
                v16h br = load_b_frag(d1T, HD, HD + w * 16, kb, lane);
                cm = wmma32(af, bm, cm);
                cr = wmma32(af, br, cr);
            }
#pragma unroll
            for (int r = 0; r < 8; ++r)
                h1h[(mt * 16 + r + 8 * hh) * HD + col] = (_Float16)(lrelu(cm[r]) + cr[r]);
        }
    }
    __syncthreads();

    // pred = leaky(h1@d2_mw+mb) + h1@d2_rw+rb   (N=78 padded to 80; 20 tiles)
    for (int t = w; t < 20; t += 8) {
        int mt = t / 5, nt = t - mt * 5;
        int col = nt * 16 + (lane & 15);
        float bm = (col < OD) ? s_m2b[col] : 0.0f;
        float br0 = (col < OD) ? s_r2b[col] : 0.0f;
        v8f cm, cr;
#pragma unroll
        for (int r = 0; r < 8; ++r) { cm[r] = bm; cr[r] = br0; }
#pragma unroll
        for (int kb = 0; kb < HD; kb += 32) {
            v16h af = load_a_frag(h1h, HD, mt * 16, kb, lane);
            v16h b1 = load_b_frag(d2T, HD, nt * 16, kb, lane);
            v16h b2 = load_b_frag(d2T, HD, 80 + nt * 16, kb, lane);
            cm = wmma32(af, b1, cm);
            cr = wmma32(af, b2, cr);
        }
#pragma unroll
        for (int r = 0; r < 8; ++r)
            pred[(mt * 16 + r + 8 * hh) * 80 + col] = lrelu(cm[r]) + cr[r];
    }
    __syncthreads();

    // cross-entropy: sum of picked log-softmax over (in,out,et) x 6 edges
    float ls = 0.0f;
    if (tid < 64) {
        const float* at = arch + (size_t)(g0 + tid) * OD;
        const float* pr = pred + tid * 80;
        const int goff[3] = {0, 4, 8};
        const int glen[3] = {4, 4, 5};
        for (int e = 0; e < 6; ++e) {
#pragma unroll
            for (int gi = 0; gi < 3; ++gi) {
                int o = e * 13 + goff[gi], L = glen[gi];
                int lab = 0; float bv = at[o];
                for (int j = 1; j < L; ++j) if (at[o + j] > bv) { bv = at[o + j]; lab = j; }
                float mx = pr[o];
                for (int j = 1; j < L; ++j) mx = fmaxf(mx, pr[o + j]);
                float se = 0.0f;
                for (int j = 0; j < L; ++j) se += __expf(pr[o + j] - mx);
                ls += pr[o + lab] - mx - __logf(se);
            }
        }
    }
    redA[tid] = ls; redB[tid] = kl;
    __syncthreads();
    for (int s = 128; s > 0; s >>= 1) {
        if (tid < s) { redA[tid] += redA[tid + s]; redB[tid] += redB[tid + s]; }
        __syncthreads();
    }
    if (tid == 0) { atomicAdd(&acc[0], redA[0]); atomicAdd(&acc[1], redB[0]); }
}

__global__ void k_init(float* acc) { acc[0] = 0.0f; acc[1] = 0.0f; }

__global__ void k_final(const float* __restrict__ acc, float* __restrict__ out)
{
    float res = -acc[0] / (float)(BATCH * 6);
    float kld = -0.5f * acc[1] / (float)(BATCH * ZD);
    out[0] = res + 0.005f * kld;
}

extern "C" void kernel_launch(void* const* d_in, const int* in_sizes, int n_in,
                              void* d_out, int out_size, void* d_ws, size_t ws_size,
                              hipStream_t stream)
{
    (void)in_sizes; (void)n_in; (void)out_size; (void)ws_size;
    const float* x     = (const float*)d_in[0];
    const float* ea    = (const float*)d_in[1];
    const float* arch  = (const float*)d_in[2];
    const float* eps   = (const float*)d_in[3];
    // d_in[4] = edge_index: fixed topology, unused
    const float* c0_rw1 = (const float*)d_in[5];
    const float* c0_rb1 = (const float*)d_in[6];
    const float* c0_rw2 = (const float*)d_in[7];
    const float* c0_rb2 = (const float*)d_in[8];
    const float* c0_kw  = (const float*)d_in[9];
    const float* c1_rw1 = (const float*)d_in[10];
    const float* c1_rb1 = (const float*)d_in[11];
    const float* c1_rw2 = (const float*)d_in[12];
    const float* c1_rb2 = (const float*)d_in[13];
    const float* c1_kw  = (const float*)d_in[14];
    const float* c2_rw1 = (const float*)d_in[15];
    const float* c2_rb1 = (const float*)d_in[16];
    const float* c2_rw2 = (const float*)d_in[17];
    const float* c2_rb2 = (const float*)d_in[18];
    const float* c2_kw  = (const float*)d_in[19];
    const float* fc3_w = (const float*)d_in[20];
    const float* fc3_b = (const float*)d_in[21];
    const float* fc4_w = (const float*)d_in[22];
    const float* fc4_b = (const float*)d_in[23];
    const float* fc5_w = (const float*)d_in[24];
    const float* fc5_b = (const float*)d_in[25];
    const float* d1_mw = (const float*)d_in[26];
    const float* d1_mb = (const float*)d_in[27];
    const float* d1_rw = (const float*)d_in[28];
    const float* d1_rb = (const float*)d_in[29];
    const float* d2_mw = (const float*)d_in[30];
    const float* d2_mb = (const float*)d_in[31];
    const float* d2_rw = (const float*)d_in[32];
    const float* d2_rb = (const float*)d_in[33];

    char* ws = (char*)d_ws;
    const size_t hbytes = (size_t)BATCH * 4 * HD * sizeof(_Float16);   // 64 MB
    _Float16* bufH = (_Float16*)ws;
    _Float16* bufR = (_Float16*)(ws + hbytes);
    float* acc = (float*)(ws + 2 * hbytes);

    dim3 blk(256);
    size_t sh_resid = (size_t)(2 * HD * HD + 2 * 64 * HD) * sizeof(_Float16);                        //  96 KB
    size_t sh_msg   = (size_t)(HD * KP + 96 * KP) * sizeof(_Float16) + (size_t)96 * 132 * sizeof(float); // ~176 KB
    size_t sh_head  = (size_t)(4 * 64 * HD + 256 * HD + 160 * HD) * sizeof(_Float16)
                    + (size_t)(64 * 64 + 64 * 80) * sizeof(float);                                   // ~204 KB

    k_init<<<dim3(1), dim3(1), 0, stream>>>(acc);
    k_conv0<<<dim3(BATCH / 2), blk, 0, stream>>>(x, ea, c0_rw1, c0_rb1, c0_rw2, c0_rb2, c0_kw, bufH);
    // conv1: resid into bufR, then msg aggregates in place into bufR
    k_resid<<<dim3(BATCH * 4 / 64), blk, sh_resid, stream>>>(bufH, c1_rw1, c1_rb1, c1_rw2, c1_rb2, bufR);
    k_msg  <<<dim3(BATCH / 16),     blk, sh_msg,   stream>>>(bufH, ea, c1_kw, bufR);
    // conv2: resid into bufH, then msg aggregates in place into bufH
    k_resid<<<dim3(BATCH * 4 / 64), blk, sh_resid, stream>>>(bufR, c2_rw1, c2_rb1, c2_rw2, c2_rb2, bufH);
    k_msg  <<<dim3(BATCH / 16),     blk, sh_msg,   stream>>>(bufR, ea, c2_kw, bufH);
    k_head <<<dim3(BATCH / 64), blk, sh_head, stream>>>(bufH, eps, arch,
        fc3_w, fc3_b, fc4_w, fc4_b, fc5_w, fc5_b,
        d1_mw, d1_mb, d1_rw, d1_rb, d2_mw, d2_mb, d2_rw, d2_rb, acc);
    k_final<<<dim3(1), dim3(1), 0, stream>>>(acc, (float*)d_out);
}
